// VQ_24696061952334
// MI455X (gfx1250) — matrix operand order repaired and checked by reading, not compiled, not run on hardware
//
#include <hip/hip_runtime.h>
#include <stdint.h>

typedef __attribute__((ext_vector_type(16))) _Float16 v16h;
typedef __attribute__((ext_vector_type(8)))  _Float16 v8h;
typedef __attribute__((ext_vector_type(8)))  float    v8f;

#define KCODES 512
#define DFEAT  64
#define HW     4096      // H*W
#define CHW    262144    // C*H*W
#define NPIX   131072    // B*H*W
// LDS codebook row stride in halfs: 144 bytes -> 16B aligned for ds_load_b128,
// 36-dword stride -> 36*r mod 64 distinct for r=0..15 (conflict-free).
#define LDS_STRIDE 72
#define SMEM_CH_BYTES (KCODES * LDS_STRIDE * 2)              // 73728 per array
#define SMEM_TOTAL    (2 * SMEM_CH_BYTES + KCODES * 4)       // 149504 bytes

#define WMMA_F16(A, B, C) \
    __builtin_amdgcn_wmma_f32_16x16x32_f16(false, (A), false, (B), (short)0, (C), false, false)

// ---------------------------------------------------------------------------
// Prep: split codebook into fp16 hi/lo (x = h + l, l = fp16(x - h)) and
// compute NEGATED half-norm  -0.5*||c||^2  per code. One block per code.
// ---------------------------------------------------------------------------
__global__ void vq_prep_kernel(const float* __restrict__ cb,
                               _Float16* __restrict__ ch,
                               _Float16* __restrict__ cl,
                               float* __restrict__ nrm) {
    int k = blockIdx.x;
    int c = threadIdx.x;
    float v = cb[k * DFEAT + c];
    _Float16 h = (_Float16)v;
    _Float16 l = (_Float16)(v - (float)h);
    ch[k * DFEAT + c] = h;
    cl[k * DFEAT + c] = l;

    __shared__ float red[DFEAT];
    red[c] = v * v;
    __syncthreads();
    if (c == 0) {
        float s = 0.f;
#pragma unroll
        for (int i = 0; i < DFEAT; ++i) s += red[i];
        nrm[k] = -0.5f * s;                 // pre-negated
    }
}

// ---------------------------------------------------------------------------
// Main: each wave owns a 16-pixel tile; loops over 512 codes in 16-code tiles,
// 2 tiles per WMMA group, ping/pong double-buffered A fragments so ds_load
// latency hides behind WMMA and no WAR hazards force v_nop fills.
// Score S[m][n] = c_m . x_n - 0.5||c_m||^2, fp16 hi/lo compensated (~2^-22).
// ---------------------------------------------------------------------------
__launch_bounds__(256)
__global__ void vq_main_kernel(const float* __restrict__ x,
                               const float* __restrict__ cb,
                               const _Float16* __restrict__ chG,
                               const _Float16* __restrict__ clG,
                               const float* __restrict__ nrmG,
                               float* __restrict__ codesOut,
                               float* __restrict__ idxOut) {
    extern __shared__ char smem[];
    _Float16* sCh  = (_Float16*)smem;
    _Float16* sCl  = (_Float16*)(smem + SMEM_CH_BYTES);
    float*    sNrm = (float*)(smem + 2 * SMEM_CH_BYTES);   // 32B aligned

    // Stage split codebook into LDS (padded rows), fully coalesced dword copies.
    {
        const uint32_t* g0 = (const uint32_t*)chG;
        const uint32_t* g1 = (const uint32_t*)clG;
        uint32_t* s0 = (uint32_t*)sCh;
        uint32_t* s1 = (uint32_t*)sCl;
        for (int i = threadIdx.x; i < KCODES * 32; i += 256) {
            int k = i >> 5, j = i & 31;
            s0[k * (LDS_STRIDE / 2) + j] = g0[i];
            s1[k * (LDS_STRIDE / 2) + j] = g1[i];
        }
        for (int i = threadIdx.x; i < KCODES; i += 256) sNrm[i] = nrmG[i];
    }
    __syncthreads();

    const int lane = threadIdx.x & 31;
    const int wave = threadIdx.x >> 5;
    const int n    = lane & 15;   // pixel within tile / N column
    const int half = lane >> 4;   // lane half selects K-subset per ISA layout

    const int pb  = blockIdx.x * 128 + wave * 16;  // global pixel tile base
    const int b   = pb >> 12;                      // batch (HW = 4096 per batch)
    const int s0p = pb & 4095;                     // spatial offset within batch
    const float* xb = x + (size_t)b * CHW + s0p;

    // B fragments (x tile, feature-major memory -> contiguous per-pixel rows).
    // B layout: elem e -> K = e + 8*(e>=8) + 8*half; feat = q*32+K.
    v16h Bh0, Bl0, Bh1, Bl1;
#pragma unroll
    for (int i = 0; i < 16; ++i) {
        float f0 = xb[(size_t)(half * 8 + i + (i >= 8 ? 8 : 0)) * HW + n];
        float f1 = xb[(size_t)(32 + half * 8 + i + (i >= 8 ? 8 : 0)) * HW + n];
        _Float16 h0 = (_Float16)f0;
        Bh0[i] = h0; Bl0[i] = (_Float16)(f0 - (float)h0);
        _Float16 h1 = (_Float16)f1;
        Bh1[i] = h1; Bl1[i] = (_Float16)(f1 - (float)h1);
    }

    float best = -3.0e38f;
    int   bidx = 0;

    // A layout: elem e -> K = e + 8*(e>=8) + 8*half; feature = q*32 + K.
    const int aoff = half * 8;

    // Load A fragments (hi/lo x 2 K-chunks) for a 2-tile group.
    auto loadGroup = [&](int ktBase, v16h (&Ah0)[2], v16h (&Ah1)[2],
                                     v16h (&Al0)[2], v16h (&Al1)[2]) {
#pragma unroll
        for (int t = 0; t < 2; ++t) {
            const _Float16* ph = sCh + (size_t)((ktBase + t) * 16 + n) * LDS_STRIDE;
            const _Float16* pl = sCl + (size_t)((ktBase + t) * 16 + n) * LDS_STRIDE;
            v8h a0 = *(const v8h*)(ph + aoff);
            v8h a1 = *(const v8h*)(ph + 16 + aoff);
            v8h a2 = *(const v8h*)(ph + 32 + aoff);
            v8h a3 = *(const v8h*)(ph + 48 + aoff);
            v8h c0 = *(const v8h*)(pl + aoff);
            v8h c1 = *(const v8h*)(pl + 16 + aoff);
            v8h c2 = *(const v8h*)(pl + 32 + aoff);
            v8h c3 = *(const v8h*)(pl + 48 + aoff);
            Ah0[t] = __builtin_shufflevector(a0, a1, 0,1,2,3,4,5,6,7,8,9,10,11,12,13,14,15);
            Ah1[t] = __builtin_shufflevector(a2, a3, 0,1,2,3,4,5,6,7,8,9,10,11,12,13,14,15);
            Al0[t] = __builtin_shufflevector(c0, c1, 0,1,2,3,4,5,6,7,8,9,10,11,12,13,14,15);
            Al1[t] = __builtin_shufflevector(c2, c3, 0,1,2,3,4,5,6,7,8,9,10,11,12,13,14,15);
        }
    };

    // 12 WMMAs as 4 round-robined chains (accH depth 2, accX depth 4, x2
    // tiles): same-accumulator ops are >=2 apart. Then running argmax.
    auto consume = [&](int ktBase, v16h (&Ah0)[2], v16h (&Ah1)[2],
                                   v16h (&Al0)[2], v16h (&Al1)[2]) {
        v8f accH[2], accX[2];
        accH[0] = *(const v8f*)(sNrm + ktBase * 16 + half * 8);
        accH[1] = *(const v8f*)(sNrm + (ktBase + 1) * 16 + half * 8);
        accX[0] = (v8f){};
        accX[1] = (v8f){};

        accH[0] = WMMA_F16(Ah0[0], Bh0, accH[0]);
        accH[1] = WMMA_F16(Ah0[1], Bh0, accH[1]);
        accX[0] = WMMA_F16(Al0[0], Bh0, accX[0]);
        accX[1] = WMMA_F16(Al0[1], Bh0, accX[1]);
        accX[0] = WMMA_F16(Ah0[0], Bl0, accX[0]);
        accX[1] = WMMA_F16(Ah0[1], Bl0, accX[1]);
        accH[0] = WMMA_F16(Ah1[0], Bh1, accH[0]);
        accH[1] = WMMA_F16(Ah1[1], Bh1, accH[1]);
        accX[0] = WMMA_F16(Al1[0], Bh1, accX[0]);
        accX[1] = WMMA_F16(Al1[1], Bh1, accX[1]);
        accX[0] = WMMA_F16(Ah1[0], Bl1, accX[0]);
        accX[1] = WMMA_F16(Ah1[1], Bl1, accX[1]);

#pragma unroll
        for (int t = 0; t < 2; ++t) {
            int rbase = (ktBase + t) * 16 + half * 8;
#pragma unroll
            for (int j = 0; j < 8; ++j) {
                float s = accH[t][j] + accX[t][j];   // dual-issue adds
                if (s > best) { best = s; bidx = rbase + j; }
            }
        }
    };

    // Ping/pong double buffer over 16 2-tile groups (32 code tiles).
    v16h pAh0[2], pAh1[2], pAl0[2], pAl1[2];
    v16h qAh0[2], qAh1[2], qAl0[2], qAl1[2];

    loadGroup(0, pAh0, pAh1, pAl0, pAl1);
    for (int kt = 0; kt < KCODES / 16; kt += 4) {
        loadGroup((kt + 2) & 31, qAh0, qAh1, qAl0, qAl1);
        consume(kt, pAh0, pAh1, pAl0, pAl1);
        loadGroup((kt + 4) & 31, pAh0, pAh1, pAl0, pAl1);  // wraps on last iter (unused)
        consume(kt + 2, qAh0, qAh1, qAl0, qAl1);
    }

    // Merge the two lane halves (same pixel, disjoint code subsets).
    {
        float ob = __shfl_xor(best, 16, 32);
        int   oi = __shfl_xor(bidx, 16, 32);
        if (ob > best || (ob == best && oi < bidx)) { best = ob; bidx = oi; }
    }

    // indices output (as float), lanes 0..15 cover the 16 pixels.
    if (half == 0) idxOut[pb + n] = (float)bidx;

    // codes output [B,C,H,W]: per feature row, 2 x 64B coalesced stores;
    // codebook gather hits the hot 128KB region in WGP$/L2.
    float* op = codesOut + (size_t)b * CHW + s0p + n;
    const float* crow = cb + (size_t)bidx * DFEAT;
#pragma unroll
    for (int i = 0; i < 32; ++i) {
        int c = i * 2 + half;
        op[(size_t)c * HW] = crow[c];
    }
}

extern "C" void kernel_launch(void* const* d_in, const int* in_sizes, int n_in,
                              void* d_out, int out_size, void* d_ws, size_t ws_size,
                              hipStream_t stream) {
    const float* x  = (const float*)d_in[0];   // [32,64,64,64] f32
    const float* cb = (const float*)d_in[1];   // [512,64] f32

    _Float16* ch  = (_Float16*)d_ws;                         // 512*64 f16 = 64KB
    _Float16* cl  = (_Float16*)((char*)d_ws + 65536);        // 64KB
    float*    nrm = (float*)((char*)d_ws + 131072);          // 2KB (negated)

    float* codesOut = (float*)d_out;                               // 8388608 floats
    float* idxOut   = codesOut + (size_t)32 * 64 * 64 * 64;        // then 131072

    vq_prep_kernel<<<KCODES, DFEAT, 0, stream>>>(cb, ch, cl, nrm);
    vq_main_kernel<<<NPIX / 128, 256, SMEM_TOTAL, stream>>>(
        x, cb, ch, cl, nrm, codesOut, idxOut);
}
